// GRU_74758200754229
// MI455X (gfx1250) — matrix operand order repaired and checked by reading, not compiled, not run on hardware
//
#include <hip/hip_runtime.h>
#include <stdint.h>

// ---------------------------------------------------------------------------
// Types for CDNA5 WMMA (wave32): v_wmma_f32_16x16x32_bf16
// ---------------------------------------------------------------------------
typedef __attribute__((ext_vector_type(16))) __bf16 v16bf;
typedef __attribute__((ext_vector_type(8)))  float  v8f;

union FragB16 { uint4 u[2]; v16bf v; };

__device__ __forceinline__ FragB16 load_bfrag(const unsigned short* p) {
  FragB16 f;
  f.u[0] = *(const uint4*)(p);
  f.u[1] = *(const uint4*)(p + 8);
  return f;
}

__device__ __forceinline__ unsigned short f32_to_bf16_rne(float f) {
  unsigned int u = __float_as_uint(f);
  u += 0x7FFFu + ((u >> 16) & 1u);          // round-to-nearest-even
  return (unsigned short)(u >> 16);
}

__device__ __forceinline__ float sigmoid_fast(float x) {
  return 1.0f / (1.0f + __expf(-x));
}
__device__ __forceinline__ float tanh_fast(float x) {
  float e = __expf(2.0f * x);
  return (e - 1.0f) / (e + 1.0f);
}

// ---------------------------------------------------------------------------
// fp32 -> bf16 elementwise conversion (grid-stride)
// ---------------------------------------------------------------------------
__global__ void cvt_f32_bf16(const float* __restrict__ in,
                             unsigned short* __restrict__ out, int n) {
  int i = blockIdx.x * blockDim.x + threadIdx.x;
  int stride = gridDim.x * blockDim.x;
  for (; i < n; i += stride) out[i] = f32_to_bf16_rne(in[i]);
}

// ---------------------------------------------------------------------------
// Input-projection GEMM:
//   C[outrow, n] = sum_k A[row, k] * W[n, k] + bias[n]
// A: [131072, K] bf16 row-major, row = b*1024 + t   (x or h1 sequence)
// W: [1536, K]   bf16 row-major ([3H, K]; B-matrix = W^T)
// C: [131072, 1536] f32, outrow = t*128 + b  (time-major for the recurrence)
// Block = 8 waves sharing one 16-row A tile staged in LDS; each wave owns a
// 16x64 output tile. B fragments double-buffered so WMMA overlaps the loads.
// ---------------------------------------------------------------------------
__global__ __launch_bounds__(256) void gemm_xproj(
    const unsigned short* __restrict__ A,
    const unsigned short* __restrict__ W,
    const float* __restrict__ bias,
    float* __restrict__ C,
    int K) {
  __shared__ __align__(16) unsigned short As[16][520];  // 16 x K bf16, padded

  const int tid  = threadIdx.x;
  const int lane = tid & 31;
  const int wave = tid >> 5;
  const int ln   = lane & 15;
  const int sel  = lane >> 4;
  const int mt   = blockIdx.x;
  const int n0   = (blockIdx.y * 8 + wave) * 64;

  // Cooperative stage of the A tile (same for all 8 waves in the block).
  const int rsh   = (K == 256) ? 5 : 6;   // uint4-chunks per row = K/8
  const int rows8 = 1 << rsh;
  for (int idx = tid; idx < (16 << rsh); idx += 256) {
    const int r = idx >> rsh, c = idx & (rows8 - 1);
    *(uint4*)(&As[r][c * 8]) =
        *(const uint4*)(A + (size_t)(mt * 16 + r) * K + c * 8);
  }
  __syncthreads();

  const v8f zero = {0.f, 0.f, 0.f, 0.f, 0.f, 0.f, 0.f, 0.f};
  v8f acc[4];
#pragma unroll
  for (int j = 0; j < 4; ++j) acc[j] = zero;

  const unsigned short* wb = W + (size_t)(n0 + ln) * K + sel * 16;
  const size_t jstride = (size_t)16 * K;

  FragB16 b0[4], b1[4];
#pragma unroll
  for (int j = 0; j < 4; ++j) b0[j] = load_bfrag(wb + j * jstride);

  const int nk = K >> 5;                  // 8 or 16 (even)
  for (int kc = 0; kc < nk; kc += 2) {
    FragB16 a;
    a.u[0] = *(const uint4*)(&As[ln][kc * 32 + sel * 16]);
    a.u[1] = *(const uint4*)(&As[ln][kc * 32 + sel * 16 + 8]);
#pragma unroll
    for (int j = 0; j < 4; ++j)
      b1[j] = load_bfrag(wb + j * jstride + (size_t)(kc + 1) * 32);
#pragma unroll
    for (int j = 0; j < 4; ++j)
      acc[j] = __builtin_amdgcn_wmma_f32_16x16x32_bf16(
          false, a.v, false, b0[j].v, (short)0, acc[j], false, false);

    FragB16 a2;
    a2.u[0] = *(const uint4*)(&As[ln][(kc + 1) * 32 + sel * 16]);
    a2.u[1] = *(const uint4*)(&As[ln][(kc + 1) * 32 + sel * 16 + 8]);
    if (kc + 2 < nk) {
#pragma unroll
      for (int j = 0; j < 4; ++j)
        b0[j] = load_bfrag(wb + j * jstride + (size_t)(kc + 2) * 32);
    }
#pragma unroll
    for (int j = 0; j < 4; ++j)
      acc[j] = __builtin_amdgcn_wmma_f32_16x16x32_bf16(
          false, a2.v, false, b1[j].v, (short)0, acc[j], false, false);
  }

#pragma unroll
  for (int j = 0; j < 4; ++j) {
    const int col = n0 + j * 16 + ln;
    const float bv = bias[col];
#pragma unroll
    for (int i = 0; i < 8; ++i) {
      const int m = mt * 16 + i + 8 * sel;                 // row = b*1024 + t
      const int outrow = ((m & 1023) << 7) | (m >> 10);    // -> t*128 + b
      C[(size_t)outrow * 1536 + col] = acc[j][i] + bv;
    }
  }
}

// ---------------------------------------------------------------------------
// GRU recurrence. Each workgroup owns a 16-row batch slice for ALL T steps;
// h state lives in LDS (fp32 + bf16 A-fragment copy), no inter-WG sync.
// Block = 8 waves. Wave w owns hidden tiles {w, w+8, w+16, w+24} and their
// z/n gate columns -> 12 accumulators. The 192-deep WMMA stream (16 K-chunks
// x 12 tiles) is software-pipelined: 4-slot B-fragment ring prefetching 4
// tiles ahead + double-buffered A fragments from LDS.
// W_hh address = base + q*65536 + g*262144 + kc*32 elements (single base ptr).
// ---------------------------------------------------------------------------
__device__ __forceinline__ constexpr int boff(int tl) {
  return (tl / 3) * 65536 + (tl % 3) * 262144;   // q*16*512*8?? -> see note
}
// note: col = (wave + 8q)*16 + 512*g + ln ; elem offset col*512 decomposes to
//       wave*8192 + q*65536 + g*262144 + ln*512  (all in bf16 elements)

__global__ __launch_bounds__(256, 1) void gru_recurrent(
    const float* __restrict__ xp,
    const unsigned short* __restrict__ Whh,
    const float* __restrict__ bhh,
    unsigned short* __restrict__ out_seq,
    float* __restrict__ h_final,
    int T) {
  __shared__ __align__(16) float          hs[16][516];  // fp32 state, padded
  __shared__ __align__(16) unsigned short hb[16][520];  // bf16 copy, padded

  const int tid  = threadIdx.x;
  const int wave = tid >> 5;
  const int lane = tid & 31;
  const int ln   = lane & 15;
  const int sel  = lane >> 4;
  const int bt   = blockIdx.x;               // batch tile: rows bt*16 .. +15

  for (int idx = tid; idx < 16 * 516; idx += 256) (&hs[0][0])[idx] = 0.0f;
  for (int idx = tid; idx < 16 * 520; idx += 256) (&hb[0][0])[idx] = 0;
  __syncthreads();

  float br[4], bz[4], bn[4];
#pragma unroll
  for (int q = 0; q < 4; ++q) {
    const int c = (wave + 8 * q) * 16 + ln;
    br[q] = bhh[c];
    bz[q] = bhh[c + 512];
    bn[q] = bhh[c + 1024];
  }

  const unsigned short* wbase =
      Whh + (size_t)wave * 8192 + (size_t)ln * 512 + sel * 16;
  const v8f zero = {0.f, 0.f, 0.f, 0.f, 0.f, 0.f, 0.f, 0.f};

  for (int t = 0; t < T; ++t) {
    v8f acc[12];
#pragma unroll
    for (int i = 0; i < 12; ++i) acc[i] = zero;

    // ---- pipelined gh = h @ Whh^T  (K = 512, 16 chunks of 32) ----
    FragB16 a_cur, a_nxt, bq[4];
    a_cur.u[0] = *(const uint4*)(&hb[ln][sel * 16]);
    a_cur.u[1] = *(const uint4*)(&hb[ln][sel * 16 + 8]);
#pragma unroll
    for (int p = 0; p < 4; ++p) bq[p] = load_bfrag(wbase + boff(p));

    for (int kc = 0; kc < 16; ++kc) {
#pragma unroll
      for (int tl = 0; tl < 12; ++tl) {
        acc[tl] = __builtin_amdgcn_wmma_f32_16x16x32_bf16(
            false, a_cur.v, false, bq[tl & 3].v, (short)0, acc[tl], false,
            false);
        // prefetch the B fragment 4 tiles ahead (wraps into next K-chunk)
        const int wrap = (tl + 4 >= 12);
        const int ptl  = wrap ? tl - 8 : tl + 4;
        if ((tl < 8) || (kc < 15)) {
          bq[tl & 3] =
              load_bfrag(wbase + boff(ptl) + (size_t)(kc + wrap) * 32);
        }
        if (tl == 5 && kc < 15) {   // next A fragment from LDS, issued early
          const int k0n = (kc + 1) << 5;
          a_nxt.u[0] = *(const uint4*)(&hb[ln][k0n + sel * 16]);
          a_nxt.u[1] = *(const uint4*)(&hb[ln][k0n + sel * 16 + 8]);
        }
      }
      a_cur = a_nxt;
    }
    __syncthreads();  // all waves done reading hb

    // ---- gate math + state update (each (m,j) owned by one lane) ----
    const float* xb = xp + (size_t)(t * 128 + bt * 16) * 1536;
#pragma unroll
    for (int q = 0; q < 4; ++q) {
      const int j = (wave + 8 * q) * 16 + ln;
#pragma unroll
      for (int i = 0; i < 8; ++i) {
        const int m = i + 8 * sel;
        const float* xr = xb + (size_t)m * 1536;
        const float xr_ = xr[j];
        const float xz_ = xr[j + 512];
        const float xn_ = xr[j + 1024];
        const float hr = acc[q * 3 + 0][i] + br[q];
        const float hz = acc[q * 3 + 1][i] + bz[q];
        const float hn = acc[q * 3 + 2][i] + bn[q];
        const float r = sigmoid_fast(xr_ + hr);
        const float z = sigmoid_fast(xz_ + hz);
        const float n = tanh_fast(xn_ + r * hn);
        const float hnew = (1.0f - z) * n + z * hs[m][j];
        hs[m][j] = hnew;
        const unsigned short hb16 = f32_to_bf16_rne(hnew);
        hb[m][j] = hb16;
        if (out_seq)
          out_seq[((size_t)(bt * 16 + m) * T + t) * 512 + j] = hb16;
      }
    }
    __syncthreads();  // new h visible before next step's fragment reads
  }

  if (h_final) {
    for (int idx = tid; idx < 16 * 512; idx += 256) {
      const int m = idx >> 9, j = idx & 511;
      h_final[(size_t)(bt * 16 + m) * 512 + j] = hs[m][j];
    }
  }
}

// ---------------------------------------------------------------------------
// out[b] = h2_last[b,:] . W_lin + b_lin   (tiny)
// ---------------------------------------------------------------------------
__global__ void final_linear(const float* __restrict__ hfin,
                             const float* __restrict__ Wlin,
                             const float* __restrict__ blin,
                             float* __restrict__ out) {
  const int b = threadIdx.x;
  if (b >= 128) return;
  float s = blin[0];
  for (int j = 0; j < 512; ++j) s += hfin[b * 512 + j] * Wlin[j];
  out[b] = s;
}

// ---------------------------------------------------------------------------
// Host launcher
// ---------------------------------------------------------------------------
extern "C" void kernel_launch(void* const* d_in, const int* in_sizes, int n_in,
                              void* d_out, int out_size, void* d_ws, size_t ws_size,
                              hipStream_t stream) {
  (void)in_sizes; (void)n_in; (void)out_size; (void)ws_size;
  const float* x     = (const float*)d_in[0];
  const float* W_ih0 = (const float*)d_in[1];
  const float* W_hh0 = (const float*)d_in[2];
  const float* b_ih0 = (const float*)d_in[3];
  const float* b_hh0 = (const float*)d_in[4];
  const float* W_ih1 = (const float*)d_in[5];
  const float* W_hh1 = (const float*)d_in[6];
  const float* b_ih1 = (const float*)d_in[7];
  const float* b_hh1 = (const float*)d_in[8];
  const float* W_lin = (const float*)d_in[9];
  const float* b_lin = (const float*)d_in[10];

  const size_t M = 131072;  // B*T = 128*1024

  char* ws = (char*)d_ws;
  size_t off = 0;
  auto take = [&](size_t bytes) -> char* {
    char* p = ws + off;
    off = (off + bytes + 255) & ~(size_t)255;
    return p;
  };
  float*          xp    = (float*)         take(M * 1536 * 4);   // 805 MB
  unsigned short* hseq  = (unsigned short*)take(M * 512 * 2);    // 134 MB
  unsigned short* xbf   = (unsigned short*)take(M * 256 * 2);    //  67 MB
  unsigned short* wbih0 = (unsigned short*)take(1536 * 256 * 2);
  unsigned short* wbhh0 = (unsigned short*)take(1536 * 512 * 2);
  unsigned short* wbih1 = (unsigned short*)take(1536 * 512 * 2);
  unsigned short* wbhh1 = (unsigned short*)take(1536 * 512 * 2);
  float*          hfin  = (float*)         take(128 * 512 * 4);

  // fp32 -> bf16 conversions
  cvt_f32_bf16<<<4096, 256, 0, stream>>>(x, xbf, (int)(M * 256));
  cvt_f32_bf16<<<1536, 256, 0, stream>>>(W_ih0, wbih0, 1536 * 256);
  cvt_f32_bf16<<<3072, 256, 0, stream>>>(W_hh0, wbhh0, 1536 * 512);
  cvt_f32_bf16<<<3072, 256, 0, stream>>>(W_ih1, wbih1, 1536 * 512);
  cvt_f32_bf16<<<3072, 256, 0, stream>>>(W_hh1, wbhh1, 1536 * 512);

  // Layer 0: input projection, then recurrence (writes bf16 h1 sequence)
  gemm_xproj<<<dim3(M / 16, 3), 256, 0, stream>>>(xbf, wbih0, b_ih0, xp, 256);
  gru_recurrent<<<8, 256, 0, stream>>>(xp, wbhh0, b_hh0, hseq, nullptr, 1024);

  // Layer 1: input projection from h1, then recurrence (keeps only final h)
  gemm_xproj<<<dim3(M / 16, 3), 256, 0, stream>>>(hseq, wbih1, b_ih1, xp, 512);
  gru_recurrent<<<8, 256, 0, stream>>>(xp, wbhh1, b_hh1, nullptr, hfin, 1024);

  // Head
  final_linear<<<1, 128, 0, stream>>>(hfin, W_lin, b_lin, (float*)d_out);
}
